// ShiftedWindowAttention_5093831213284
// MI455X (gfx1250) — matrix operand order repaired and checked by reading, not compiled, not run on hardware
//
#include <hip/hip_runtime.h>
#include <hip/hip_bf16.h>

typedef __bf16 bf16_t;
typedef bf16_t v16bf __attribute__((ext_vector_type(16)));
typedef float  v8f   __attribute__((ext_vector_type(8)));
typedef unsigned short u16;
typedef unsigned int   u32;

#define WH    7
#define N_TOK 49
#define HEADS 4
#define DIM   128
#define IMG   112
#define SHF   3

// ---------- helpers ----------

__device__ __forceinline__ u16 f2bf(float f) {
    // round-to-nearest-even f32 -> bf16
    u32 u = __float_as_uint(f);
    u32 r = u + 0x7FFFu + ((u >> 16) & 1u);
    return (u16)(r >> 16);
}

// A fragment (16x32 bf16, M x K): lane (m = lane&15, half = lane>>4)
// vec[i] <-> k = kbase + 16*half + i
__device__ __forceinline__ v16bf frag_a(const u16* row, int kbase, int hf) {
    union { uint4 q[2]; v16bf v; } u;
    u.q[0] = *(const uint4*)(row + kbase + 16 * hf);
    u.q[1] = *(const uint4*)(row + kbase + 16 * hf + 8);
    return u.v;
}

// B fragment (32x16 bf16, K x N) sourced from an N-major [n][k] array:
// lane (n = lane&15, khalf = lane>>4); vec[i] <-> k = kbase + 16*khalf + i
__device__ __forceinline__ v16bf frag_b(const u16* row, int kbase, int khalf) {
    union { uint4 q[2]; v16bf v; } u;
    const u16* p = row + kbase + 16 * khalf;
    u.q[0] = *(const uint4*)(p);
    u.q[1] = *(const uint4*)(p + 8);
    return u.v;
}

__device__ __forceinline__ v8f wmma_bf16(v16bf a, v16bf b, v8f c) {
    return __builtin_amdgcn_wmma_f32_16x16x32_bf16(false, a, false, b, (short)0, c, false, false);
}

// ---------- prep: pack weights to bf16, fold q-scale, precompute rpb+mask ----------

__global__ void swin_prep_kernel(
    const float* __restrict__ qw, const float* __restrict__ kw,
    const float* __restrict__ vw, const float* __restrict__ pw,
    const float* __restrict__ qb, const float* __restrict__ kb,
    const float* __restrict__ vb, const float* __restrict__ pb,
    const float* __restrict__ table,
    u16* __restrict__ wqkv, u16* __restrict__ projw,
    float* __restrict__ bqkv, float* __restrict__ bproj,
    float* __restrict__ biasall)
{
    const float scale = 0.17677669529663687f; // 32^-0.5 folded into Wq, bq
    int gid = blockIdx.x * blockDim.x + threadIdx.x;
    int stride = gridDim.x * blockDim.x;

    // combined QKV weight, N-major [384][128] bf16
    for (int idx = gid; idx < 384 * 128; idx += stride) {
        int n = idx >> 7, k = idx & 127;
        float w;
        if (n < 128)      w = qw[n * 128 + k] * scale;
        else if (n < 256) w = kw[(n - 128) * 128 + k];
        else              w = vw[(n - 256) * 128 + k];
        wqkv[idx] = f2bf(w);
    }
    // proj weight, N-major [128][128] bf16 (proj_weight is already [out][in])
    for (int idx = gid; idx < 128 * 128; idx += stride) projw[idx] = f2bf(pw[idx]);

    for (int idx = gid; idx < 384; idx += stride) {
        float bsv;
        if (idx < 128)      bsv = qb[idx] * scale;
        else if (idx < 256) bsv = kb[idx - 128];
        else                bsv = vb[idx - 256];
        bqkv[idx] = bsv;
    }
    for (int idx = gid; idx < 128; idx += stride) bproj[idx] = pb[idx];

    // biasall[wt][h][i][j] = rel-pos-bias + shift mask; wt = (bottom?2:0)+(right?1:0)
    for (int idx = gid; idx < 4 * 4 * 49 * 49; idx += stride) {
        int j  = idx % 49;
        int i  = (idx / 49) % 49;
        int h  = (idx / (49 * 49)) % 4;
        int wt = idx / (4 * 49 * 49);
        int ih = i / 7, iw = i % 7, jh = j / 7, jw = j % 7;
        int rel = (ih - jh + 6) * 13 + (iw - jw + 6);
        float r = table[rel * 4 + h];
        int eh = wt >> 1, ew = wt & 1;
        int ri = (eh ? (ih < 4 ? 1 : 2) : 0) * 3 + (ew ? (iw < 4 ? 1 : 2) : 0);
        int rj = (eh ? (jh < 4 ? 1 : 2) : 0) * 3 + (ew ? (jw < 4 ? 1 : 2) : 0);
        float mk = (ri != rj) ? -100.0f : 0.0f;
        biasall[idx] = r + mk;
    }
}

// ---------- fused per-window attention: one workgroup per window ----------

__global__ __launch_bounds__(256) void swin_attn_kernel(
    const float* __restrict__ x,
    const u16* __restrict__ wqkv,      // [384][128] bf16, N-major
    const u16* __restrict__ projw,     // [128][128] bf16, N-major
    const float* __restrict__ bqkv,    // [384]
    const float* __restrict__ bproj,   // [128]
    const float* __restrict__ biasall, // [4][4][49][49]
    float* __restrict__ out)
{
    __shared__ alignas(16) u16   Xbf[64 * 128];       // window tokens, bf16
    // one contiguous QKV buffer: sec 0 = Q [token][dim], sec 1 = K [token][dim],
    // sec 2 = V stored transposed [dim][token]; each head slice is 2048 u16
    __shared__ alignas(16) u16   QKVs[3 * HEADS * 2048];
    __shared__ alignas(16) float Sbuf[64 * 64];       // attn logits/probs f32
    __shared__ alignas(16) u16   Pbf[64 * 64];        // probs bf16
    __shared__ alignas(16) u16   Obf[64 * 128];       // attention output bf16

    const int blk  = blockIdx.x;
    const int b    = blk >> 8;
    const int wi   = (blk >> 4) & 15;
    const int wj   = blk & 15;
    const int tid  = threadIdx.x;
    const int lane = tid & 31;
    const int wave = tid >> 5;
    const int lm   = lane & 15;
    const int hf   = lane >> 4;

    if (tid == 0) { __builtin_prefetch(wqkv, 0, 0); __builtin_prefetch(projw, 0, 0); }

    // ---- stage 1: gather rolled window, f32 -> bf16 into LDS (pad rows 49..63 = 0)
    for (int e = 0; e < 8; ++e) {
        int idx = e * 256 + tid;      // 2048 float4 slots = 64 rows x 32
        int m   = idx >> 5;
        int c4  = idx & 31;
        u16* dst = &Xbf[m * 128 + c4 * 4];
        if (m < N_TOK) {
            int i = m / WH, j = m % WH;
            int gh = wi * WH + i + SHF; if (gh >= IMG) gh -= IMG;
            int gw = wj * WH + j + SHF; if (gw >= IMG) gw -= IMG;
            const float4 v = *(const float4*)(x + (((size_t)b * IMG + gh) * IMG + gw) * DIM + c4 * 4);
            dst[0] = f2bf(v.x); dst[1] = f2bf(v.y); dst[2] = f2bf(v.z); dst[3] = f2bf(v.w);
        } else {
            dst[0] = 0; dst[1] = 0; dst[2] = 0; dst[3] = 0;
        }
    }
    // zero P once: pad rows/cols stay zero across all heads
    {
        uint4 z = {0u, 0u, 0u, 0u};
        uint4* p = (uint4*)Pbf;       // 64*64 u16 = 512 uint4
        p[tid] = z;
        p[tid + 256] = z;
    }
    __syncthreads();

    // ---- stage 2: QKV GEMM  Y[64][384] = Xbf @ Wqkv^T + b   (96 tiles, K=4 steps)
    for (int it = 0; it < 12; ++it) {
        int t  = wave + it * 8;      // 0..95
        int mt = t / 24, nt = t % 24;
        v8f acc = {};
        const u16* arow = &Xbf[(mt * 16 + lm) * 128];
        const u16* brow = wqkv + (size_t)(nt * 16 + lm) * 128;
        for (int ks = 0; ks < 4; ++ks) {
            v16bf a  = frag_a(arow, ks * 32, hf);
            v16bf bb = frag_b(brow, ks * 32, hf);
            acc = wmma_bf16(a, bb, acc);
        }
        int n = nt * 16 + lm;
        float bias = bqkv[n];
        int sec = n >> 7;            // 0=q 1=k 2=v
        int nn  = n & 127;
        int hh  = nn >> 5;
        int d   = nn & 31;
        // pure-arithmetic destination (no branches -> v_cndmask):
        int isV     = sec >> 1;                    // 1 only for V
        int rowoff  = isV ? d * 64 : d;            // V is [dim][token]
        int mstride = isV ? 1 : 32;                // Q/K are [token][dim]
        u16* base = &QKVs[(sec * HEADS + hh) * 2048 + rowoff];
        int m0 = mt * 16 + 8 * hf;
        for (int r = 0; r < 8; ++r)
            base[(m0 + r) * mstride] = f2bf(acc[r] + bias);
    }
    __syncthreads();

    // ---- stage 3: per-head attention
    const int wtype = ((wi == 15) ? 2 : 0) + ((wj == 15) ? 1 : 0);
    const int srow = tid >> 2;        // softmax row, 4 lanes per row
    const int sl   = tid & 3;         // lane within quad
    for (int h = 0; h < HEADS; ++h) {
        const u16* Qh = &QKVs[(0 * HEADS + h) * 2048];  // [token][dim], ld 32
        const u16* Kh = &QKVs[(1 * HEADS + h) * 2048];  // [token][dim], ld 32
        const u16* Vh = &QKVs[(2 * HEADS + h) * 2048];  // [dim][token], ld 64

        // S = Q K^T : 16 tiles, single K=32 step (head_dim == WMMA K!)
        for (int t = 0; t < 2; ++t) {
            int tt = wave + t * 8;   // 0..15
            int mt = tt >> 2, nt = tt & 3;
            v8f acc = {};
            v16bf a  = frag_a(Qh + (mt * 16 + lm) * 32, 0, hf);
            v16bf bb = frag_b(Kh + (nt * 16 + lm) * 32, 0, hf); // B = K^T (N-major)
            acc = wmma_bf16(a, bb, acc);
            for (int r = 0; r < 8; ++r)
                Sbuf[(mt * 16 + r + 8 * hf) * 64 + nt * 16 + lm] = acc[r];
        }
        __syncthreads();

        // softmax: 4 lanes per row, quad reductions via shfl_xor
        if (srow < N_TOK) {
            const float* bp = biasall + (((size_t)wtype * 4 + h) * 49 + srow) * 49;
            float* srw = &Sbuf[srow * 64];
            float mx = -1e30f;
            for (int j = sl; j < N_TOK; j += 4) {
                float s = srw[j] + bp[j];
                srw[j] = s;
                mx = fmaxf(mx, s);
            }
            mx = fmaxf(mx, __shfl_xor(mx, 1, 4));
            mx = fmaxf(mx, __shfl_xor(mx, 2, 4));
            float sum = 0.f;
            for (int j = sl; j < N_TOK; j += 4) {
                float e2 = __expf(srw[j] - mx);
                srw[j] = e2;
                sum += e2;
            }
            sum += __shfl_xor(sum, 1, 4);
            sum += __shfl_xor(sum, 2, 4);
            float inv = 1.0f / sum;
            for (int j = sl; j < N_TOK; j += 4)
                Pbf[srow * 64 + j] = f2bf(srw[j] * inv);
        }
        __syncthreads();

        // O_h = P @ V : 8 tiles, K = 2 steps of 32
        {
            int mt = wave >> 1, nt = wave & 1;
            v8f acc = {};
            const u16* arow = &Pbf[(mt * 16 + lm) * 64];
            const u16* brow = Vh + (nt * 16 + lm) * 64;   // Vt [dim][token] = N-major B
            for (int ks = 0; ks < 2; ++ks) {
                v16bf a  = frag_a(arow, ks * 32, hf);
                v16bf bb = frag_b(brow, ks * 32, hf);
                acc = wmma_bf16(a, bb, acc);
            }
            for (int r = 0; r < 8; ++r)
                Obf[(mt * 16 + r + 8 * hf) * 128 + h * 32 + nt * 16 + lm] = f2bf(acc[r]);
        }
        __syncthreads();
    }

    // ---- stage 4: projection + scatter with inverse roll
    for (int it = 0; it < 4; ++it) {
        int t  = wave + it * 8;     // 0..31
        int mt = t >> 3, nt = t & 7;
        v8f acc = {};
        const u16* arow = &Obf[(mt * 16 + lm) * 128];
        const u16* brow = projw + (size_t)(nt * 16 + lm) * 128;
        for (int ks = 0; ks < 4; ++ks) {
            v16bf a  = frag_a(arow, ks * 32, hf);
            v16bf bb = frag_b(brow, ks * 32, hf);
            acc = wmma_bf16(a, bb, acc);
        }
        int n = nt * 16 + lm;
        float bias = bproj[n];
        for (int r = 0; r < 8; ++r) {
            int m = mt * 16 + r + 8 * hf;
            if (m < N_TOK) {
                int i = m / WH, j = m % WH;
                int gh = wi * WH + i + SHF; if (gh >= IMG) gh -= IMG;
                int gw = wj * WH + j + SHF; if (gw >= IMG) gw -= IMG;
                out[(((size_t)b * IMG + gh) * IMG + gw) * DIM + n] = acc[r] + bias;
            }
        }
    }
}

// ---------- launch ----------

extern "C" void kernel_launch(void* const* d_in, const int* in_sizes, int n_in,
                              void* d_out, int out_size, void* d_ws, size_t ws_size,
                              hipStream_t stream) {
    const float* x  = (const float*)d_in[0];
    // d_in[1], d_in[2] (input_k/input_v) are unused by the reference
    const float* qw = (const float*)d_in[3];
    const float* kw = (const float*)d_in[4];
    const float* vw = (const float*)d_in[5];
    const float* pw = (const float*)d_in[6];
    const float* qb = (const float*)d_in[7];
    const float* kb = (const float*)d_in[8];
    const float* vb = (const float*)d_in[9];
    const float* pb = (const float*)d_in[10];
    const float* tb = (const float*)d_in[11];

    char* ws = (char*)d_ws;
    u16*   wqkv    = (u16*)(ws + 0);        // 384*128*2   = 98304
    u16*   projw   = (u16*)(ws + 98304);    // 128*128*2   = 32768
    float* bqkv    = (float*)(ws + 131072); // 384*4       = 1536
    float* bproj   = (float*)(ws + 132608); // 128*4       = 512
    float* biasall = (float*)(ws + 133120); // 4*4*49*49*4 = 153664
    float* out = (float*)d_out;

    swin_prep_kernel<<<dim3(128), dim3(256), 0, stream>>>(
        qw, kw, vw, pw, qb, kb, vb, pb, tb, wqkv, projw, bqkv, bproj, biasall);
    swin_attn_kernel<<<dim3(8192), dim3(256), 0, stream>>>(
        x, wqkv, projw, bqkv, bproj, biasall, out);
}
